// Multiplication_64939905516354
// MI455X (gfx1250) — matrix operand order repaired and checked by reading, not compiled, not run on hardware
//
#include <hip/hip_runtime.h>

// ---- types ------------------------------------------------------------
typedef __attribute__((ext_vector_type(16))) __bf16        v16bf;
typedef __attribute__((ext_vector_type(8)))  float         v8f;
typedef __attribute__((ext_vector_type(8)))  unsigned short u16x8;
typedef __attribute__((ext_vector_type(4)))  float         f32x4;

static constexpr int Q = 16384;  // queries
static constexpr int C = 2048;   // classes
static constexpr int S = 5;      // supports per class
static constexpr int D = 512;    // feature dim

__device__ __forceinline__ unsigned short f2bf(float f) {
  // round-to-nearest-even f32 -> bf16
  unsigned u = __float_as_uint(f);
  return (unsigned short)((u + 0x7FFFu + ((u >> 16) & 1u)) >> 16);
}

union FragBF {                     // WMMA 16x32 bf16 operand = 16 halfs/lane
  struct { u16x8 lo; u16x8 hi; } p;
  v16bf v;
};

__device__ __forceinline__ v8f wmma_bf16(const FragBF& a, const FragBF& b, v8f c) {
  return __builtin_amdgcn_wmma_f32_16x16x32_bf16(
      false, a.v, false, b.v, (short)0, c, false, false);
}

// ---- kernel 1: x f32 -> bf16 (8 elems / thread) -----------------------
__global__ void cvt_x_bf16(const float* __restrict__ x,
                           unsigned short* __restrict__ xb) {
  const int t = blockIdx.x * blockDim.x + threadIdx.x;   // one u16x8 chunk
  const f32x4* src = (const f32x4*)x + (size_t)t * 2;
  f32x4 a = src[0], b = src[1];
  u16x8 o;
  o[0] = f2bf(a[0]); o[1] = f2bf(a[1]); o[2] = f2bf(a[2]); o[3] = f2bf(a[3]);
  o[4] = f2bf(b[0]); o[5] = f2bf(b[1]); o[6] = f2bf(b[2]); o[7] = f2bf(b[3]);
  ((u16x8*)xb)[t] = o;
}

// ---- kernel 2: prototypes = mean over S, store bf16 row-major [C][D] --
__global__ void proto_bf16(const float* __restrict__ y,
                           unsigned short* __restrict__ proto) {
  const int t = blockIdx.x * blockDim.x + threadIdx.x;   // C*D/8 threads
  const int c = t >> 6;          // D/8 = 64 chunks per class row
  const int d = (t & 63) * 8;
  float acc[8] = {0.f, 0.f, 0.f, 0.f, 0.f, 0.f, 0.f, 0.f};
#pragma unroll
  for (int s = 0; s < S; ++s) {
    const f32x4* p = (const f32x4*)(y + ((size_t)(c * S + s) * D + d));
    f32x4 a = p[0], b = p[1];
    acc[0] += a[0]; acc[1] += a[1]; acc[2] += a[2]; acc[3] += a[3];
    acc[4] += b[0]; acc[5] += b[1]; acc[6] += b[2]; acc[7] += b[3];
  }
  u16x8 o;
#pragma unroll
  for (int j = 0; j < 8; ++j) o[j] = f2bf(acc[j] * 0.2f);
  ((u16x8*)proto)[t] = o;
}

// ---- kernel 3: fused bf16-WMMA GEMM + row softmax ---------------------
// One workgroup (8 wave32) owns 32 query rows x all 2048 classes.
// LDS: logits 32*2048 f32 (256KB) + A tile 32*512 bf16 (32KB) = 288KB
// (fits CDNA5's 320KB-per-WGP LDS; logits never touch global memory).
// Each wave register-blocks 2Mx2N: 4 accumulators per K-step fed by
// 2 A-fragments (LDS) + 2 B-fragments (global/L2) -> 4 loads : 4 wmma.
__global__ __launch_bounds__(256) void proto_gemm_softmax(
    const unsigned short* __restrict__ xb,
    const unsigned short* __restrict__ proto,
    float* __restrict__ out) {
  extern __shared__ char smem[];
  float*          logits = (float*)smem;                                // [32][C]
  unsigned short* aTile  = (unsigned short*)(smem + 32 * C * sizeof(float)); // [32][D]

  const int tid     = threadIdx.x;
  const int lane    = tid & 31;
  const int wave    = tid >> 5;
  const int rowBase = blockIdx.x * 32;

  // stage A tile: 32 rows x 512 bf16 = 32KB, cooperative b128 copies
  {
    const u16x8* src = (const u16x8*)(xb + (size_t)rowBase * D);
    u16x8*       dst = (u16x8*)aTile;
    for (int i = tid; i < 32 * D / 8; i += 256) dst[i] = src[i];
  }
  __syncthreads();

  const int half16 = lane >> 4;      // lane group 0/1 (ISA 16-bit layouts)
  const int l16    = lane & 15;
  const int g8     = half16 * 8;     // A: K-offset 0 / 8 within 32-chunk
  const int g16    = half16 * 16;    // unused; B loads use g8 too
  // Two M-tile A rows per wave (M 0..15 and 16..31 of this row block)
  const unsigned short* Arow0 = aTile + l16 * D;
  const unsigned short* Arow1 = Arow0 + 16 * D;

  // 64 N-tile pairs of 32 columns; 8 pairs per wave
  for (int ntp = wave; ntp < C / 32; ntp += 8) {
    const int cbase = ntp * 32;
    const unsigned short* Brow0 = proto + (size_t)(cbase + l16) * D;
    const unsigned short* Brow1 = Brow0 + 16 * D;
    // pull next pair's prototype rows toward the WGP early (L2 -> L0)
    __builtin_prefetch(Brow0 + (size_t)8 * 32 * D, 0, 1);
    __builtin_prefetch(Brow1 + (size_t)8 * 32 * D, 0, 1);

    v8f acc00 = {}, acc01 = {}, acc10 = {}, acc11 = {};
#pragma unroll 2
    for (int kk = 0; kk < D; kk += 32) {
      FragBF a0, a1, b0, b1;
      // A 16x32: lanes 0-15 hold K[0..7],K[16..23]; lanes 16-31 K[8..15],K[24..31]
      a0.p.lo = *(const u16x8*)(Arow0 + kk + g8);
      a0.p.hi = *(const u16x8*)(Arow0 + kk + 16 + g8);
      a1.p.lo = *(const u16x8*)(Arow1 + kk + g8);
      a1.p.hi = *(const u16x8*)(Arow1 + kk + 16 + g8);
      // B 32x16: lanes 0-15 hold K[0..7],K[16..23]; lanes 16-31 K[8..15],K[24..31] of column l16
      const u16x8* bp0 = (const u16x8*)(Brow0 + kk + g8);
      const u16x8* bp1 = (const u16x8*)(Brow1 + kk + g8);
      b0.p.lo = bp0[0]; b0.p.hi = bp0[2];
      b1.p.lo = bp1[0]; b1.p.hi = bp1[2];
      acc00 = wmma_bf16(a0, b0, acc00);
      acc01 = wmma_bf16(a0, b1, acc01);
      acc10 = wmma_bf16(a1, b0, acc10);
      acc11 = wmma_bf16(a1, b1, acc11);
    }
    // C/D f32 layout: VGPR i -> M = i + (lane>=16 ? 8 : 0), N = lane%16
    const int col0 = cbase + l16;
    const int col1 = col0 + 16;
    const int rb0  = half16 * 8;        // M-tile 0 rows
    const int rb1  = 16 + half16 * 8;   // M-tile 1 rows
#pragma unroll
    for (int i = 0; i < 8; ++i) {
      logits[(rb0 + i) * C + col0] = acc00[i];
      logits[(rb0 + i) * C + col1] = acc01[i];
      logits[(rb1 + i) * C + col0] = acc10[i];
      logits[(rb1 + i) * C + col1] = acc11[i];
    }
  }
  __syncthreads();

  // softmax straight out of LDS: each wave owns 4 rows
  for (int rr = 0; rr < 4; ++rr) {
    const int r = wave * 4 + rr;
    const float* Lrow = logits + r * C;
    float m = -3.4e38f;
    for (int c2 = lane; c2 < C; c2 += 32) m = fmaxf(m, Lrow[c2]);
#pragma unroll
    for (int off = 16; off > 0; off >>= 1) m = fmaxf(m, __shfl_xor(m, off, 32));
    float s = 0.f;
    for (int c2 = lane; c2 < C; c2 += 32) s += __expf(Lrow[c2] - m);
#pragma unroll
    for (int off = 16; off > 0; off >>= 1) s += __shfl_xor(s, off, 32);
    const float inv = 1.0f / s;
    float* Orow = out + (size_t)(rowBase + r) * C;
    for (int c2 = lane; c2 < C; c2 += 32) Orow[c2] = __expf(Lrow[c2] - m) * inv;
  }
}

// ---- host launcher ----------------------------------------------------
extern "C" void kernel_launch(void* const* d_in, const int* in_sizes, int n_in,
                              void* d_out, int out_size, void* d_ws, size_t ws_size,
                              hipStream_t stream) {
  const float* x = (const float*)d_in[0];   // [Q][D] f32
  const float* y = (const float*)d_in[1];   // [C][S][D] f32
  float* out = (float*)d_out;               // [Q][C] f32

  unsigned short* xb    = (unsigned short*)d_ws;      // Q*D bf16 (16 MB)
  unsigned short* proto = xb + (size_t)Q * D;         // C*D bf16 (2 MB)

  cvt_x_bf16<<<(Q * D / 8) / 256, 256, 0, stream>>>(x, xb);
  proto_bf16<<<(C * D / 8) / 256, 256, 0, stream>>>(y, proto);

  const size_t smemBytes =
      (size_t)32 * C * sizeof(float) + (size_t)32 * D * sizeof(unsigned short);
  (void)hipFuncSetAttribute((const void*)proto_gemm_softmax,
                            hipFuncAttributeMaxDynamicSharedMemorySize,
                            (int)smemBytes);
  proto_gemm_softmax<<<Q / 32, 256, smemBytes, stream>>>(xb, proto, out);
}
